// SelectiveSSM_57509612094214
// MI455X (gfx1250) — matrix operand order repaired and checked by reading, not compiled, not run on hardware
//
#include <hip/hip_runtime.h>
#include <cstdint>
#include <cstddef>

// ---------------- gfx1250 feature probes ----------------
#if defined(__has_builtin)
#  if __has_builtin(__builtin_amdgcn_global_load_async_to_lds_b128) && \
      __has_builtin(__builtin_amdgcn_s_wait_asynccnt)
#    define HAVE_ASYNC_LDS 1
#  else
#    define HAVE_ASYNC_LDS 0
#  endif
#else
#  define HAVE_ASYNC_LDS 0
#endif

#define AS1 __attribute__((address_space(1)))
#define AS3 __attribute__((address_space(3)))

typedef __attribute__((ext_vector_type(16))) __bf16 v16bf;
typedef __attribute__((ext_vector_type(8)))  __bf16 bf16x8;
typedef __attribute__((ext_vector_type(8)))  float  v8f;
typedef __attribute__((ext_vector_type(4)))  int    v4i;

union V16 { v16bf v; bf16x8 h[2]; };

// Problem constants (reference file)
static constexpr int DM   = 768;     // D_MODEL
static constexpr int DI   = 1536;    // D_INNER
static constexpr int DST  = 8;       // D_STATE
static constexpr int LSEQ = 4096;    // SEQ
static constexpr int MROWS = 8192;   // BATCH*SEQ

__device__ __forceinline__ unsigned short f2bf(float f) {
  unsigned int u = __float_as_uint(f);
  u += 0x7FFFu + ((u >> 16) & 1u);           // round-to-nearest-even
  return (unsigned short)(u >> 16);
}

// ---------------- elementwise f32 -> bf16 cast ----------------
__global__ __launch_bounds__(256) void cast_bf16_kernel(const float* __restrict__ src,
                                                        unsigned short* __restrict__ dst, int n) {
  int i = blockIdx.x * 256 + threadIdx.x;
  if (i < n) dst[i] = f2bf(src[i]);
}

// ---------------- transpose + cast: src[R x C] f32 -> dst[C x R] bf16 ----------------
__global__ __launch_bounds__(256) void transpose_bf16_kernel(const float* __restrict__ src,
                                                             unsigned short* __restrict__ dst,
                                                             int R, int C) {
  int i = blockIdx.x * 256 + threadIdx.x;
  if (i >= R * C) return;
  int r = i / C, c = i % C;
  dst[(size_t)c * R + r] = f2bf(src[i]);
}

// ---------------- WMMA bf16 GEMM ----------------
// C[M x N] = A[M x K](bf16,row-major) * B[K x N], with B supplied pre-transposed
// as Bt[N x K](bf16,row-major).  Block tile 64(M) x 256(N), 8 waves (2x4),
// each wave 32x64 = 2x4 v_wmma_f32_16x16x32_bf16 tiles, K-step 32.
// A tiles are double-buffered in LDS via async global->LDS copies so the DMA
// for tile k+1 overlaps the 8 WMMAs of tile k (s_wait_asynccnt 1).
// splitCol>0: column split store (xz -> x_b | z), else plain store to C0.
__global__ __launch_bounds__(256) void gemm_bf16_wmma_kernel(
    const unsigned short* __restrict__ A, const unsigned short* __restrict__ Bt,
    float* __restrict__ C0, float* __restrict__ C1,
    int M, int N, int K, int splitCol) {
  __shared__ __align__(16) unsigned short ldsA[2][64 * 32];

  const int tid  = threadIdx.x;
  const int wid  = tid >> 5;
  const int lane = tid & 31;
  const int hlf  = lane >> 4;       // wave32 half select
  const int l15  = lane & 15;
  const int mw   = wid & 1;         // 2 M-waves
  const int nw   = wid >> 1;        // 4 N-waves
  const int mBase = blockIdx.x * 64;
  const int nBase = blockIdx.y * 256;

  // A staging: 256 threads x 16B = 64 rows x 32 cols bf16
  const int aRow = tid >> 2;
  const int aCol = (tid & 3) * 8;

  auto stage = [&](int k0, int buf) {
    const unsigned short* gsrc = A + (size_t)(mBase + aRow) * K + k0 + aCol;
#if HAVE_ASYNC_LDS
    __builtin_amdgcn_global_load_async_to_lds_b128(
        (AS1 v4i*)const_cast<unsigned short*>(gsrc),
        (AS3 v4i*)(&ldsA[buf][aRow * 32 + aCol]), 0, 0);
#else
    uint4 tmp = *reinterpret_cast<const uint4*>(gsrc);
    *reinterpret_cast<uint4*>(&ldsA[buf][aRow * 32 + aCol]) = tmp;
#endif
  };

  v8f acc[2][4] = {};

  stage(0, 0);                                   // prologue: tile 0 in flight
  for (int k0 = 0; k0 < K; k0 += 32) {
    const int cur = (k0 >> 5) & 1;
    if (k0 + 32 < K) {
      stage(k0 + 32, cur ^ 1);                   // next tile streams behind compute
#if HAVE_ASYNC_LDS
      __builtin_amdgcn_s_wait_asynccnt(1);       // current tile resident (in-order)
#endif
    } else {
#if HAVE_ASYNC_LDS
      __builtin_amdgcn_s_wait_asynccnt(0);
#endif
    }
    __syncthreads();

    // A fragments from LDS (layout: elem e <-> K = 16*(e>>3) + 8*half + (e&7))
    V16 afr[2];
#pragma unroll
    for (int mi = 0; mi < 2; ++mi) {
      int row = mw * 32 + mi * 16 + l15;
      afr[mi].h[0] = *reinterpret_cast<const bf16x8*>(&ldsA[cur][row * 32 + 8 * hlf]);
      afr[mi].h[1] = *reinterpret_cast<const bf16x8*>(&ldsA[cur][row * 32 + 16 + 8 * hlf]);
    }
    // B fragments straight from pre-transposed global (elem e <-> K = 16*(e>>3) + 8*half + (e&7))
    V16 bfr[4];
#pragma unroll
    for (int ni = 0; ni < 4; ++ni) {
      int col = nBase + nw * 64 + ni * 16 + l15;
      const bf16x8* bp = reinterpret_cast<const bf16x8*>(Bt + (size_t)col * K + k0 + 8 * hlf);
      bfr[ni].h[0] = bp[0];
      bfr[ni].h[1] = bp[2];
      // unconditional -> straight-line global_prefetch_b8 (speculative, drop-safe)
      __builtin_prefetch(Bt + (size_t)col * K + (k0 + 32) + 16 * hlf, 0, 1);
    }
#pragma unroll
    for (int mi = 0; mi < 2; ++mi)
#pragma unroll
      for (int ni = 0; ni < 4; ++ni)
        acc[mi][ni] = __builtin_amdgcn_wmma_f32_16x16x32_bf16(
            false, afr[mi].v, false, bfr[ni].v, (short)0, acc[mi][ni], false, false);

    __syncthreads();                             // all reads of buf `cur` done before reuse
  }

  // Epilogue: D/C layout: VGPR r -> M = 8*half + r, N = l15
#pragma unroll
  for (int mi = 0; mi < 2; ++mi) {
    int rowB = mBase + mw * 32 + mi * 16 + 8 * hlf;
#pragma unroll
    for (int ni = 0; ni < 4; ++ni) {
      int col = nBase + nw * 64 + ni * 16 + l15;
#pragma unroll
      for (int r = 0; r < 8; ++r) {
        float v = acc[mi][ni][r];
        int rr = rowB + r;
        if (splitCol > 0) {
          if (col < splitCol) C0[(size_t)rr * splitCol + col] = v;
          else                C1[(size_t)rr * splitCol + (col - splitCol)] = v;
        } else {
          C0[(size_t)rr * N + col] = v;
        }
      }
    }
  }
}

// ---------------- depthwise causal conv (width 4) + SiLU ----------------
__global__ __launch_bounds__(256) void conv_silu_kernel(const float* __restrict__ xb,
                                                        const float* __restrict__ conv_w,
                                                        const float* __restrict__ conv_b,
                                                        float* __restrict__ xc) {
  int i = blockIdx.x * 256 + threadIdx.x;          // i = t*DI + d
  if (i >= MROWS * DI) return;
  int t = i / DI, d = i % DI;
  int seq = t & (LSEQ - 1);
  float acc = conv_b[d];
#pragma unroll
  for (int j = 0; j < 4; ++j) {
    int s = seq - 3 + j;
    if (s >= 0) acc += conv_w[d * 4 + j] * xb[(size_t)(t - 3 + j) * DI + d];
  }
  xc[i] = acc / (1.0f + __expf(-acc));             // SiLU
}

// ---------------- x_c @ W_x (K=1536, 17 outputs): one wave per row ----------------
__global__ __launch_bounds__(256) void proj_kernel(const float* __restrict__ xc,
                                                   const float* __restrict__ Wx,
                                                   float* __restrict__ Bp, float* __restrict__ Cp,
                                                   float* __restrict__ dtp) {
  int row  = blockIdx.x * 8 + (threadIdx.x >> 5);
  int lane = threadIdx.x & 31;
  float acc[17];
#pragma unroll
  for (int c = 0; c < 17; ++c) acc[c] = 0.0f;
  const float* xr = xc + (size_t)row * DI;
  for (int d = lane; d < DI; d += 32) {
    float u = xr[d];
    const float* w = Wx + d * 17;
#pragma unroll
    for (int c = 0; c < 17; ++c) acc[c] += u * w[c];
  }
#pragma unroll
  for (int off = 16; off >= 1; off >>= 1)
#pragma unroll
    for (int c = 0; c < 17; ++c) acc[c] += __shfl_xor(acc[c], off, 32);
  if (lane == 0) {
#pragma unroll
    for (int c = 0; c < DST; ++c) { Bp[row * DST + c] = acc[c]; Cp[row * DST + c] = acc[DST + c]; }
    dtp[row] = acc[16];
  }
}

// ---------------- fused selective scan + D skip + silu(z) gate -> bf16 ----------------
// One lane per (batch, channel): 3072 sequential scans, coalesced across d.
__global__ __launch_bounds__(256) void scan_gate_kernel(
    const float* __restrict__ xc, const float* __restrict__ zb,
    const float* __restrict__ Bp, const float* __restrict__ Cp, const float* __restrict__ dtp,
    const float* __restrict__ W_dt, const float* __restrict__ b_dt,
    const float* __restrict__ A_log, const float* __restrict__ D_skip,
    unsigned short* __restrict__ g_bf) {
  int idx = blockIdx.x * 256 + threadIdx.x;        // 0 .. 3071
  int b = idx / DI, d = idx % DI;
  float A[DST], h[DST];
#pragma unroll
  for (int s = 0; s < DST; ++s) { A[s] = -__expf(A_log[d * DST + s]); h[s] = 0.0f; }
  const float wdt = W_dt[d], bdt = b_dt[d], dsk = D_skip[d];
  const int base = b * LSEQ;
  for (int t = 0; t < LSEQ; ++t) {
    int row = base + t;
    float u  = xc[(size_t)row * DI + d];
    float xv = dtp[row] * wdt + bdt;
    float dt = (xv > 20.0f) ? xv : log1pf(__expf(xv));   // softplus
    float y = 0.0f;
#pragma unroll
    for (int s = 0; s < DST; ++s) {
      float dA = fminf(__expf(dt * A[s]), 1.0f);
      h[s] = dA * h[s] + dt * Bp[row * DST + s] * u;
      y += h[s] * Cp[row * DST + s];
    }
    y += u * dsk;
    float zv = zb[(size_t)row * DI + d];
    float g  = y * (zv / (1.0f + __expf(-zv)));          // * silu(z)
    g_bf[(size_t)row * DI + d] = f2bf(g);
  }
}

// ---------------- launcher ----------------
extern "C" void kernel_launch(void* const* d_in, const int* in_sizes, int n_in,
                              void* d_out, int out_size, void* d_ws, size_t ws_size,
                              hipStream_t stream) {
  (void)in_sizes; (void)n_in; (void)out_size; (void)ws_size;
  const float* x      = (const float*)d_in[0];
  const float* W_in   = (const float*)d_in[1];
  const float* conv_w = (const float*)d_in[2];
  const float* conv_b = (const float*)d_in[3];
  const float* W_x    = (const float*)d_in[4];
  const float* W_dt   = (const float*)d_in[5];
  const float* b_dt   = (const float*)d_in[6];
  const float* A_log  = (const float*)d_in[7];
  const float* D_skip = (const float*)d_in[8];
  const float* W_out  = (const float*)d_in[9];
  float* out = (float*)d_out;

  char* ws = (char*)d_ws;
  size_t off = 0;
  auto alloc = [&](size_t bytes) -> char* {
    char* p = ws + off;
    off += (bytes + 255) & ~(size_t)255;
    return p;
  };
  unsigned short* x_bf   = (unsigned short*)alloc((size_t)MROWS * DM * 2);
  unsigned short* wt_in  = (unsigned short*)alloc((size_t)(2 * DI) * DM * 2); // [3072 x 768]
  unsigned short* wt_out = (unsigned short*)alloc((size_t)DM * DI * 2);       // [768 x 1536]
  float* xb  = (float*)alloc((size_t)MROWS * DI * 4);
  float* zb  = (float*)alloc((size_t)MROWS * DI * 4);
  float* xc  = (float*)alloc((size_t)MROWS * DI * 4);
  float* Bp  = (float*)alloc((size_t)MROWS * DST * 4);
  float* Cp  = (float*)alloc((size_t)MROWS * DST * 4);
  float* dtp = (float*)alloc((size_t)MROWS * 4);
  unsigned short* g_bf = (unsigned short*)alloc((size_t)MROWS * DI * 2);

  // 1) precision casts / weight transposes (bf16, B supplied as [N x K])
  cast_bf16_kernel<<<(MROWS * DM) / 256, 256, 0, stream>>>(x, x_bf, MROWS * DM);
  transpose_bf16_kernel<<<(DM * 2 * DI) / 256, 256, 0, stream>>>(W_in, wt_in, DM, 2 * DI);
  transpose_bf16_kernel<<<(DI * DM) / 256, 256, 0, stream>>>(W_out, wt_out, DI, DM);

  // 2) xz = x @ W_in  -> split into x_b | z          (M=8192, N=3072, K=768)
  gemm_bf16_wmma_kernel<<<dim3(MROWS / 64, (2 * DI) / 256), 256, 0, stream>>>(
      x_bf, wt_in, xb, zb, MROWS, 2 * DI, DM, DI);

  // 3) causal depthwise conv + SiLU
  conv_silu_kernel<<<(MROWS * DI) / 256, 256, 0, stream>>>(xb, conv_w, conv_b, xc);

  // 4) ssm = x_c @ W_x -> B_p, C_p, dt_p
  proj_kernel<<<MROWS / 8, 256, 0, stream>>>(xc, W_x, Bp, Cp, dtp);

  // 5) selective scan + D skip + silu(z) gate, emit bf16 activation
  scan_gate_kernel<<<(2 * DI) / 256, 256, 0, stream>>>(xc, zb, Bp, Cp, dtp,
                                                       W_dt, b_dt, A_log, D_skip, g_bf);

  // 6) out = g @ W_out                               (M=8192, N=768, K=1536)
  gemm_bf16_wmma_kernel<<<dim3(MROWS / 64, DM / 256), 256, 0, stream>>>(
      g_bf, wt_out, out, nullptr, MROWS, DM, DI, 0);
}
